// SNNLanguageModel_39256001085801
// MI455X (gfx1250) — matrix operand order repaired and checked by reading, not compiled, not run on hardware
//
#include <hip/hip_runtime.h>
#include <hip/hip_bf16.h>

// ---------------- model constants ----------------
#define BB   4
#define SS   16
#define KK   8
#define DD   512
#define DNX  1024
#define LL   4
#define DFF  1536
#define VOC  6144
#define MR   16          // WMMA M tile (rows 4..15 are zero padding)

typedef __bf16 bf16;
typedef bf16  bf16x8  __attribute__((ext_vector_type(8)));
typedef bf16  bf16x16 __attribute__((ext_vector_type(16)));
typedef float f32x4   __attribute__((ext_vector_type(4)));
typedef float f32x8   __attribute__((ext_vector_type(8)));

__device__ __forceinline__ float sigf(float x)      { return 1.0f / (1.0f + __expf(-x)); }
__device__ __forceinline__ float softplusf(float x) { return (x > 20.0f) ? x : log1pf(__expf(x)); }

__device__ __forceinline__ f32x8 zero8() {
    f32x8 z;
#pragma unroll
    for (int r = 0; r < 8; ++r) z[r] = 0.0f;
    return z;
}

// ---- WMMA K-chain: acc += A[16,Kd] @ W[N,Kd]^T tile at column n0..n0+15 ----
// A fragment (m = lane&15, half = lane>>4): k in [kt+half*8, +8) U [kt+16+half*8, +8)
// B fragment (n = n0 + (lane&15)):          k in [kt+half*8, +8) U [kt+16+half*8, +8)
__device__ __forceinline__ f32x8 mm_chain(const bf16* __restrict__ A, const bf16* __restrict__ W,
                                          int Kd, int lane, int ncol, f32x8 acc) {
    const int half = lane >> 4, mrow = lane & 15;
    for (int kt = 0; kt < Kd; kt += 32) {
        const bf16* Arow = A + (size_t)mrow * Kd + kt + half * 8;
        bf16x8 a0 = *(const bf16x8*)(Arow);
        bf16x8 a1 = *(const bf16x8*)(Arow + 16);
        const bf16* Wrow = W + (size_t)ncol * Kd + kt + half * 8;
        bf16x8 b0 = *(const bf16x8*)(Wrow);
        bf16x8 b1 = *(const bf16x8*)(Wrow + 16);
        bf16x16 a, b;
#pragma unroll
        for (int j = 0; j < 8; ++j) { a[j] = a0[j]; a[8 + j] = a1[j]; b[j] = b0[j]; b[8 + j] = b1[j]; }
        acc = __builtin_amdgcn_wmma_f32_16x16x32_bf16(false, a, false, b, (short)0, acc, false, false);
    }
    return acc;
}

// f32 A-operand variant (per-lane row pointer; padding lanes point at a zeroed row)
__device__ __forceinline__ f32x8 mm_chain_f32(const float* __restrict__ Arow,
                                              const bf16* __restrict__ W, int Kd,
                                              int lane, int ncol, f32x8 acc) {
    const int half = lane >> 4;
    for (int kt = 0; kt < Kd; kt += 32) {
        const float* p0 = Arow + kt + half * 8;
        f32x4 x0 = *(const f32x4*)(p0);
        f32x4 x1 = *(const f32x4*)(p0 + 4);
        f32x4 x2 = *(const f32x4*)(p0 + 16);
        f32x4 x3 = *(const f32x4*)(p0 + 20);
        const bf16* Wrow = W + (size_t)ncol * Kd + kt + half * 8;
        bf16x8 b0 = *(const bf16x8*)(Wrow);
        bf16x8 b1 = *(const bf16x8*)(Wrow + 16);
        bf16x16 a, b;
#pragma unroll
        for (int j = 0; j < 4; ++j) {
            a[j] = (bf16)x0[j]; a[4 + j] = (bf16)x1[j];
            a[8 + j] = (bf16)x2[j]; a[12 + j] = (bf16)x3[j];
        }
#pragma unroll
        for (int j = 0; j < 8; ++j) { b[j] = b0[j]; b[8 + j] = b1[j]; }
        acc = __builtin_amdgcn_wmma_f32_16x16x32_bf16(false, a, false, b, (short)0, acc, false, false);
    }
    return acc;
}

#define TILE_PROLOGUE(Nn)                                          \
    const int lane = threadIdx.x & 31;                             \
    const int tile = blockIdx.x * (blockDim.x >> 5) + (threadIdx.x >> 5); \
    const int n0 = tile * 16;                                      \
    if (n0 >= (Nn)) return;                                        \
    const int half = lane >> 4;                                    \
    const int ncol = n0 + (lane & 15);

// ---------------- fused layer kernels ----------------
// (1) state-space block: I/beta/alpha/vth GEMM chains + gate + soft-reset spike
__global__ __launch_bounds__(256) void k_blockA(
    const bf16* __restrict__ spk, const bf16* __restrict__ Vr,
    const bf16* __restrict__ Wi,  const bf16* __restrict__ Wbx,
    const bf16* __restrict__ Wax, const bf16* __restrict__ Wtx,
    const bf16* __restrict__ WbV, const bf16* __restrict__ WaV, const bf16* __restrict__ WtV,
    const float* __restrict__ bb, const float* __restrict__ ba, const float* __restrict__ bt,
    float* __restrict__ Vst, bf16* __restrict__ Vw, bf16* __restrict__ sBf) {
    TILE_PROLOGUE(DNX)
    f32x8 cI = mm_chain(spk, Wi, DD, lane, ncol, zero8());
    f32x8 cB = mm_chain(Vr, WbV, DNX, lane, ncol, mm_chain(spk, Wbx, DD, lane, ncol, zero8()));
    f32x8 cA = mm_chain(Vr, WaV, DNX, lane, ncol, mm_chain(spk, Wax, DD, lane, ncol, zero8()));
    f32x8 cT = mm_chain(Vr, WtV, DNX, lane, ncol, mm_chain(spk, Wtx, DD, lane, ncol, zero8()));
    const float bbc = bb[ncol], bac = ba[ncol], btc = bt[ncol];
#pragma unroll
    for (int r = 0; r < 8; ++r) {
        int m = r + half * 8;
        size_t i = (size_t)m * DNX + ncol;
        float beta  = sigf(cB[r] + bbc);
        float alpha = sigf(cA[r] + bac);
        float vth   = 0.1f + softplusf(cT[r] + btc);
        float Vn    = beta * Vst[i] + alpha * cI[r];
        float sp    = (Vn >= vth) ? 1.0f : 0.0f;
        Vn -= sp * vth;                       // soft reset
        Vst[i] = Vn;
        Vw[i]  = (bf16)Vn;
        sBf[i] = (bf16)sp;
    }
}

// (2) y = (s@Wo)*sigmoid(spk@Wgt) + spk@Wsk ; PLIF -> sb
__global__ __launch_bounds__(256) void k_blockY(
    const bf16* __restrict__ sBf, const bf16* __restrict__ spk,
    const bf16* __restrict__ Wo, const bf16* __restrict__ Wgt, const bf16* __restrict__ Wsk,
    float* __restrict__ ub, bf16* __restrict__ sb,
    const float* __restrict__ wblk, int l, float vth) {
    TILE_PROLOGUE(DD)
    f32x8 c1 = mm_chain(sBf, Wo, DNX, lane, ncol, zero8());
    f32x8 c2 = mm_chain(spk, Wgt, DD, lane, ncol, zero8());
    f32x8 c3 = mm_chain(spk, Wsk, DD, lane, ncol, zero8());
    const float w = sigf(wblk[l]);
#pragma unroll
    for (int r = 0; r < 8; ++r) {
        int m = r + half * 8;
        size_t i = (size_t)m * DD + ncol;
        float y  = c1[r] * sigf(c2[r]) + c3[r];
        float u  = ub[i] + w * (y - ub[i]);
        float sp = (u >= vth) ? 1.0f : 0.0f;
        ub[i] = u * (1.0f - sp);
        sb[i] = (bf16)sp;
    }
}

// (3) SwiGLU front: g/u GEMMs + dual PLIF (vth=1) + product
__global__ __launch_bounds__(256) void k_ffn1(
    const bf16* __restrict__ sb,
    const bf16* __restrict__ Wg, const bf16* __restrict__ Wu,
    float* __restrict__ ug, float* __restrict__ uu,
    const float* __restrict__ wg, const float* __restrict__ wu, int l,
    bf16* __restrict__ m) {
    TILE_PROLOGUE(DFF)
    f32x8 cG = mm_chain(sb, Wg, DD, lane, ncol, zero8());
    f32x8 cU = mm_chain(sb, Wu, DD, lane, ncol, zero8());
    const float a = sigf(wg[l]), b = sigf(wu[l]);
#pragma unroll
    for (int r = 0; r < 8; ++r) {
        int mm = r + half * 8;
        size_t i = (size_t)mm * DFF + ncol;
        float u1 = ug[i] + a * (cG[r] - ug[i]);
        float s1 = (u1 >= 1.0f) ? 1.0f : 0.0f;
        ug[i] = u1 * (1.0f - s1);
        float u2 = uu[i] + b * (cU[r] - uu[i]);
        float s2 = (u2 >= 1.0f) ? 1.0f : 0.0f;
        uu[i] = u2 * (1.0f - s2);
        m[i] = (bf16)(s1 * s2);
    }
}

// (4) h = m@Wd + sb@Wfs ; PLIF -> sf ; (last layer) decode accumulate
// decMode: 0 = none, 1 = dec = bitw*s (first frame), 2 = dec += bitw*s
__global__ __launch_bounds__(256) void k_ffn2(
    const bf16* __restrict__ m, const bf16* __restrict__ sb,
    const bf16* __restrict__ Wd, const bf16* __restrict__ Wfs,
    float* __restrict__ uf, const float* __restrict__ wf, int l, float vth,
    bf16* __restrict__ sf, float* __restrict__ dec, float bitw, int decMode) {
    TILE_PROLOGUE(DD)
    f32x8 c = mm_chain(m, Wd, DFF, lane, ncol, zero8());
    c = mm_chain(sb, Wfs, DD, lane, ncol, c);
    const float w = sigf(wf[l]);
#pragma unroll
    for (int r = 0; r < 8; ++r) {
        int mm = r + half * 8;
        size_t i = (size_t)mm * DD + ncol;
        float u  = uf[i] + w * (c[r] - uf[i]);
        float sp = (u >= vth) ? 1.0f : 0.0f;
        uf[i] = u * (1.0f - sp);
        sf[i] = (bf16)sp;
        if (decMode == 1)      dec[i] = bitw * sp;
        else if (decMode == 2) dec[i] += bitw * sp;
    }
}

// encoder: gather embedding rows as f32 A operand, GEMM vs enc_W, sigmoid+bias,
// and emit all KK MSB-first bit frames in the epilogue
__global__ __launch_bounds__(256) void k_encgemm(
    const int* __restrict__ tok, const float* __restrict__ embed,
    const float* __restrict__ zrow, const bf16* __restrict__ Wenc,
    const float* __restrict__ enc_b, bf16* __restrict__ frames, int s) {
    TILE_PROLOGUE(DD)
    const int mrow = lane & 15;
    const float* Arow = (mrow < BB) ? (embed + (size_t)tok[mrow * SS + s] * DD) : zrow;
    f32x8 c = mm_chain_f32(Arow, Wenc, DD, lane, ncol, zero8());
    const float eb = enc_b[ncol];
#pragma unroll
    for (int r = 0; r < 8; ++r) {
        int m = r + half * 8;
        float x = sigf(c[r] + eb);
#pragma unroll
        for (int k = 0; k < KK; ++k) {
            float bit = (x >= 0.5f) ? 1.0f : 0.0f;
            frames[(size_t)k * MR * DD + (size_t)m * DD + ncol] = (bf16)bit;
            x = (x - 0.5f * bit) * 2.0f;
        }
    }
}

// decoder GEMM: zz_pre = dec(f32) @ dec_W^T
__global__ __launch_bounds__(256) void k_decgemm(
    const float* __restrict__ dec, const bf16* __restrict__ Wdec, float* __restrict__ zz) {
    TILE_PROLOGUE(DD)
    const int mrow = lane & 15;
    f32x8 c = mm_chain_f32(dec + (size_t)mrow * DD, Wdec, DD, lane, ncol, zero8());
#pragma unroll
    for (int r = 0; r < 8; ++r) zz[(size_t)(r + half * 8) * DD + ncol] = c[r];
}

// logits GEMM: out[b,s,:] = zz @ embed^T (writes d_out directly, rows < BB only)
__global__ __launch_bounds__(256) void k_logits(
    const bf16* __restrict__ zz, const bf16* __restrict__ Wemb,
    float* __restrict__ out, int s) {
    TILE_PROLOGUE(VOC)
    f32x8 c = mm_chain(zz, Wemb, DD, lane, ncol, zero8());
#pragma unroll
    for (int r = 0; r < 8; ++r) {
        int m = r + half * 8;
        if (m < BB) out[((size_t)m * SS + s) * VOC + ncol] = c[r];
    }
}

// ---------------- small utility kernels ----------------
__global__ void k_f2bf(const float* __restrict__ s, bf16* __restrict__ d, int n) {
    for (int i = blockIdx.x * blockDim.x + threadIdx.x; i < n; i += gridDim.x * blockDim.x)
        d[i] = (bf16)s[i];
}

__global__ void k_zero(float* __restrict__ p, int n) {
    for (int i = blockIdx.x * blockDim.x + threadIdx.x; i < n; i += gridDim.x * blockDim.x)
        p[i] = 0.0f;
}

// RMS-norm over D per row; writes bf16 A operand for the logits GEMM
__global__ __launch_bounds__(256) void k_rmsnorm(const float* __restrict__ zz,
                                                 const float* __restrict__ dec_b,
                                                 const float* __restrict__ nw,
                                                 bf16* __restrict__ out) {
    __shared__ float red[256];
    int row = blockIdx.x;
    int tid = threadIdx.x;
    float ss = 0.0f;
    for (int c = tid; c < DD; c += 256) {
        float x = zz[row * DD + c] + dec_b[c];
        ss += x * x;
    }
    red[tid] = ss;
    __syncthreads();
    for (int st = 128; st > 0; st >>= 1) {
        if (tid < st) red[tid] += red[tid + st];
        __syncthreads();
    }
    float inv = rsqrtf(red[0] / (float)DD + 1e-6f);
    for (int c = tid; c < DD; c += 256) {
        float x = zz[row * DD + c] + dec_b[c];
        out[row * DD + c] = (bf16)(x * inv * nw[c]);
    }
}

// ---------------- host sequencing ----------------
extern "C" void kernel_launch(void* const* d_in, const int* in_sizes, int n_in,
                              void* d_out, int out_size, void* d_ws, size_t ws_size,
                              hipStream_t stream) {
    (void)in_sizes; (void)n_in; (void)out_size; (void)ws_size;

    const int*   tok     = (const int*)  d_in[0];
    const float* embed   = (const float*)d_in[1];
    const float* enc_W   = (const float*)d_in[2];
    const float* enc_b   = (const float*)d_in[3];
    const float* dec_W   = (const float*)d_in[4];
    const float* dec_b   = (const float*)d_in[5];
    const float* norm_w  = (const float*)d_in[6];
    const float* W_in    = (const float*)d_in[7];
    const float* Wbx     = (const float*)d_in[8];
    const float* Wax     = (const float*)d_in[9];
    const float* Wtx     = (const float*)d_in[10];
    const float* WbV     = (const float*)d_in[11];
    const float* WaV     = (const float*)d_in[12];
    const float* WtV     = (const float*)d_in[13];
    const float* b_beta  = (const float*)d_in[14];
    const float* b_alpha = (const float*)d_in[15];
    const float* b_th    = (const float*)d_in[16];
    const float* W_gate  = (const float*)d_in[17];
    const float* W_skip  = (const float*)d_in[18];
    const float* W_out   = (const float*)d_in[19];
    const float* w_blk   = (const float*)d_in[20];
    const float* Wg      = (const float*)d_in[21];
    const float* Wu      = (const float*)d_in[22];
    const float* Wd      = (const float*)d_in[23];
    const float* Wfs     = (const float*)d_in[24];
    const float* w_g     = (const float*)d_in[25];
    const float* w_u     = (const float*)d_in[26];
    const float* w_f     = (const float*)d_in[27];
    float* out = (float*)d_out;

    // ---- workspace layout (16B-aligned slabs) ----
    char*  ws  = (char*)d_ws;
    size_t off = 0;
    auto allocB = [&](size_t n) { bf16*  p = (bf16*) (ws + off); off += ((n * sizeof(bf16)  + 15) & ~15ull); return p; };
    auto allocF = [&](size_t n) { float* p = (float*)(ws + off); off += ((n * sizeof(float) + 15) & ~15ull); return p; };

    // bf16 weight copies (~77MB -> entirely L2 resident on MI455X's 192MB L2)
    bf16* wEnc = allocB((size_t)DD * DD);
    bf16* wDec = allocB((size_t)DD * DD);
    bf16* wEmb = allocB((size_t)VOC * DD);
    bf16* wIn  = allocB((size_t)LL * DNX * DD);
    bf16* wBx  = allocB((size_t)LL * DNX * DD);
    bf16* wAx  = allocB((size_t)LL * DNX * DD);
    bf16* wTx  = allocB((size_t)LL * DNX * DD);
    bf16* wBV  = allocB((size_t)LL * DNX * DNX);
    bf16* wAV  = allocB((size_t)LL * DNX * DNX);
    bf16* wTV  = allocB((size_t)LL * DNX * DNX);
    bf16* wGt  = allocB((size_t)LL * DD * DD);
    bf16* wSk  = allocB((size_t)LL * DD * DD);
    bf16* wWo  = allocB((size_t)LL * DD * DNX);
    bf16* wWg  = allocB((size_t)LL * DFF * DD);
    bf16* wWu  = allocB((size_t)LL * DFF * DD);
    bf16* wWd  = allocB((size_t)LL * DD * DFF);
    bf16* wWfs = allocB((size_t)LL * DD * DD);

    // activations
    bf16* frames = allocB((size_t)KK * MR * DD);
    bf16* sfBuf  = allocB((size_t)MR * DD);
    bf16* sBf    = allocB((size_t)MR * DNX);
    bf16* sbBf   = allocB((size_t)MR * DD);
    bf16* mBf    = allocB((size_t)MR * DFF);
    bf16* zzBf   = allocB((size_t)MR * DD);
    float* fDec  = allocF((size_t)MR * DD);
    float* fZZ   = allocF((size_t)MR * DD);

    // recurrent states + zero row (contiguous -> single zero-fill)
    char*  stateBase = ws + off;
    float* Vst   = allocF((size_t)LL * MR * DNX);
    float* ubS   = allocF((size_t)LL * MR * DD);
    float* ugS   = allocF((size_t)LL * MR * DFF);
    float* uuS   = allocF((size_t)LL * MR * DFF);
    float* ufS   = allocF((size_t)LL * MR * DD);
    bf16*  VstBf = allocB((size_t)2 * LL * MR * DNX);   // double-buffered by frame parity
    float* zrow  = allocF((size_t)DD);                  // zero row for padded A lanes
    size_t stateFloats = (size_t)(ws + off - stateBase) / sizeof(float);

    const float blkv[LL] = {0.3f, 0.05f, 0.05f, 0.05f};
    const float ffnv[LL] = {0.5f, 0.15f, 0.15f, 0.15f};

#define CVT(src, dst, n) k_f2bf<<<dim3((unsigned)(((n) + 255) / 256)), dim3(256), 0, stream>>>((src), (dst), (int)(n))
#define GRID(Nn) dim3((unsigned)((Nn) / 128)), dim3(256), 0, stream

    // ---- per-call init: weights -> bf16, states -> 0 ----
    CVT(enc_W,  wEnc, (size_t)DD * DD);
    CVT(dec_W,  wDec, (size_t)DD * DD);
    CVT(embed,  wEmb, (size_t)VOC * DD);
    CVT(W_in,   wIn,  (size_t)LL * DNX * DD);
    CVT(Wbx,    wBx,  (size_t)LL * DNX * DD);
    CVT(Wax,    wAx,  (size_t)LL * DNX * DD);
    CVT(Wtx,    wTx,  (size_t)LL * DNX * DD);
    CVT(WbV,    wBV,  (size_t)LL * DNX * DNX);
    CVT(WaV,    wAV,  (size_t)LL * DNX * DNX);
    CVT(WtV,    wTV,  (size_t)LL * DNX * DNX);
    CVT(W_gate, wGt,  (size_t)LL * DD * DD);
    CVT(W_skip, wSk,  (size_t)LL * DD * DD);
    CVT(W_out,  wWo,  (size_t)LL * DD * DNX);
    CVT(Wg,     wWg,  (size_t)LL * DFF * DD);
    CVT(Wu,     wWu,  (size_t)LL * DFF * DD);
    CVT(Wd,     wWd,  (size_t)LL * DD * DFF);
    CVT(Wfs,    wWfs, (size_t)LL * DD * DD);
    k_zero<<<dim3((unsigned)((stateFloats + 255) / 256)), dim3(256), 0, stream>>>(
        (float*)stateBase, (int)stateFloats);

    // ---- sequential token loop ----
    for (int s = 0; s < SS; ++s) {
        k_encgemm<<<GRID(DD)>>>(tok, embed, zrow, wEnc, enc_b, frames, s);

        for (int k = 0; k < KK; ++k) {
            const float bitw = ldexpf(1.0f, -(k + 1));
            const size_t rb = (size_t)(k & 1) * LL * MR * DNX;        // read buffer
            const size_t wb = (size_t)((k + 1) & 1) * LL * MR * DNX;  // write buffer
            for (int l = 0; l < LL; ++l) {
                const bf16* spk = (l == 0) ? frames + (size_t)k * MR * DD : sfBuf;
                const size_t lDN = (size_t)l * MR * DNX;

                k_blockA<<<GRID(DNX)>>>(
                    spk, VstBf + rb + lDN,
                    wIn + (size_t)l * DNX * DD, wBx + (size_t)l * DNX * DD,
                    wAx + (size_t)l * DNX * DD, wTx + (size_t)l * DNX * DD,
                    wBV + (size_t)l * DNX * DNX, wAV + (size_t)l * DNX * DNX,
                    wTV + (size_t)l * DNX * DNX,
                    b_beta + (size_t)l * DNX, b_alpha + (size_t)l * DNX, b_th + (size_t)l * DNX,
                    Vst + lDN, VstBf + wb + lDN, sBf);

                k_blockY<<<GRID(DD)>>>(
                    sBf, spk,
                    wWo + (size_t)l * DD * DNX, wGt + (size_t)l * DD * DD, wSk + (size_t)l * DD * DD,
                    ubS + (size_t)l * MR * DD, sbBf, w_blk, l, blkv[l]);

                k_ffn1<<<GRID(DFF)>>>(
                    sbBf, wWg + (size_t)l * DFF * DD, wWu + (size_t)l * DFF * DD,
                    ugS + (size_t)l * MR * DFF, uuS + (size_t)l * MR * DFF,
                    w_g, w_u, l, mBf);

                const int decMode = (l == LL - 1) ? ((k == 0) ? 1 : 2) : 0;
                k_ffn2<<<GRID(DD)>>>(
                    mBf, sbBf, wWd + (size_t)l * DD * DFF, wWfs + (size_t)l * DD * DD,
                    ufS + (size_t)l * MR * DD, w_f, l, ffnv[l],
                    sfBuf, fDec, bitw, decMode);
            }
        }

        k_decgemm<<<GRID(DD)>>>(fDec, wDec, fZZ);
        k_rmsnorm<<<dim3(MR), dim3(256), 0, stream>>>(fZZ, dec_b, norm_w, zzBf);
        k_logits<<<GRID(VOC)>>>(zzBf, wEmb, out, s);
    }

#undef CVT
#undef GRID
}
